// TemporalClassificationModel_65712999629030
// MI455X (gfx1250) — matrix operand order repaired and checked by reading, not compiled, not run on hardware
//
#include <hip/hip_runtime.h>
#include <hip/hip_bf16.h>

typedef _Float16 half_t;
typedef __attribute__((ext_vector_type(16))) _Float16 v16h;
typedef __attribute__((ext_vector_type(8)))  float    v8f;

#define DDIM  768
#define LSEQ  512
#define BATCH 64
#define NROWS (BATCH * LSEQ)            // 32768
#define NTIME 12
#define NVOC  30522
#define NFILT 5000
#define BLD   ((size_t)NROWS * DDIM)    // 25165824 elements per output tensor
#define NWELT (NTIME * DDIM * DDIM)     // 7077888 elements per weight tensor
#define BMWORDS ((NVOC + 31) / 32)      // 954

// ---------------- helpers ----------------
__device__ __forceinline__ void store4h(half_t* dst, float x, float y, float z, float w) {
  half_t h[4] = {(half_t)x, (half_t)y, (half_t)z, (half_t)w};
  uint2 u;
  __builtin_memcpy(&u, h, 8);
  *(uint2*)dst = u;
}

// Branch-free tanh: v_exp_f32 + v_rcp_f32, clamped so no inf/NaN handling needed.
__device__ __forceinline__ float fast_tanh(float x) {
  float cx = fminf(fmaxf(x, -9.0f), 9.0f);
  float e  = __expf(2.0f * cx);
  return (e - 1.0f) * __builtin_amdgcn_rcpf(e + 1.0f);
}

// Direct global -> LDS async copy (GLOBAL_LOAD_ASYNC_TO_LDS_B128, ASYNCcnt),
// GVS mode: saddr = uniform 64-bit base (SGPR pair), vaddr = 32-bit byte offset,
// vdst = LDS byte address. No data VGPRs are used.
__device__ __forceinline__ void async_copy_b128(unsigned lds_addr, unsigned gvoff,
                                                const void* sbase) {
  asm volatile("global_load_async_to_lds_b128 %0, %1, %2"
               :
               : "v"(lds_addr), "v"(gvoff), "s"(sbase)
               : "memory");
}
__device__ __forceinline__ void wait_asynccnt0() {
  asm volatile("s_wait_asynccnt 0x0" ::: "memory");
}

// ---------------- prep kernels ----------------
__global__ void k_zero_bitmap(unsigned int* bm) {
  int i = blockIdx.x * blockDim.x + threadIdx.x;
  if (i < BMWORDS) bm[i] = 0u;
}

__global__ void k_scatter_filter(const int* __restrict__ vf, unsigned int* __restrict__ bm) {
  int i = blockIdx.x * blockDim.x + threadIdx.x;
  if (i < NFILT) {
    unsigned int v = (unsigned int)vf[i];
    atomicOr(&bm[v >> 5], 1u << (v & 31));
  }
}

__global__ void k_row_filter(const int* __restrict__ reviews, const unsigned int* __restrict__ bm,
                             float* __restrict__ filt) {
  int r = blockIdx.x * blockDim.x + threadIdx.x;
  if (r < NROWS) {
    unsigned int tok = (unsigned int)reviews[r];
    filt[r] = (float)((bm[tok >> 5] >> (tok & 31)) & 1u);
  }
}

// Transpose + convert weights once: W[t][k][n] f32  ->  Wt[t][n][k] f16.
__global__ __launch_bounds__(256) void k_transpose_weights(
    const float* __restrict__ W1, const float* __restrict__ W2,
    half_t* __restrict__ W1t, half_t* __restrict__ W2t) {
  __shared__ float tile[32][33];
  const int tx = threadIdx.x;        // 0..31
  const int ty = threadIdx.y;        // 0..7
  const int z  = blockIdx.z;         // 0..2*NTIME-1
  const int t  = z % NTIME;
  const float*  src = (z < NTIME ? W1 : W2) + (size_t)t * DDIM * DDIM;
  half_t*       dst = (z < NTIME ? W1t : W2t) + (size_t)t * DDIM * DDIM;
  const int r0 = blockIdx.y * 32;
  const int c0 = blockIdx.x * 32;
#pragma unroll
  for (int i = 0; i < 4; ++i)
    tile[ty + 8 * i][tx] = src[(size_t)(r0 + ty + 8 * i) * DDIM + c0 + tx];
  __syncthreads();
#pragma unroll
  for (int i = 0; i < 4; ++i)
    dst[(size_t)(c0 + ty + 8 * i) * DDIM + r0 + tx] = (half_t)tile[tx][ty + 8 * i];
}

__global__ void k_gather_embs(const int* __restrict__ reviews, const float* __restrict__ emb,
                              half_t* __restrict__ Xh) {
  size_t i4 = ((size_t)blockIdx.x * blockDim.x + threadIdx.x) * 4;
  if (i4 < BLD) {
    int row = (int)(i4 / DDIM);
    int col = (int)(i4 % DDIM);
    int tok = reviews[row];
    float4 e = *(const float4*)(emb + (size_t)tok * DDIM + col);
    store4h(Xh + i4, e.x, e.y, e.z, e.w);
  }
}

// ---------------- fused GEMM kernel ----------------
// C = A[NROWS,D] @ W[t]  with W pre-transposed as Wt[t][n][k].
// STAGE 1: H = tanh(C + bias[t])                  -> Hout (f16)
// STAGE 2: Y = (C + bias[t]) * filt[row]          -> Out[PASS*BLD + ...]
//          PASS==1 also writes input_embs = emb_gather + Y at Out[2*BLD + ...]
// PASS 0: t = max(times[b]-1, 0)   PASS 1: t = times[b]
template <int STAGE, int PASS>
__global__ __launch_bounds__(256) void gemm_kernel(
    const half_t* __restrict__ A,      // [NROWS, DDIM] f16
    const half_t* __restrict__ Wt,     // [NTIME, DDIM(n), DDIM(k)] f16 (transposed)
    const float* __restrict__ bias,    // [NTIME, DDIM] f32
    const int* __restrict__ times,     // [BATCH]
    const float* __restrict__ filt,    // [NROWS]
    const int* __restrict__ reviews,   // [NROWS]
    const float* __restrict__ emb,     // [NVOC, DDIM]
    half_t* __restrict__ Hout,         // [NROWS, DDIM] f16
    float* __restrict__ Out)           // d_out base (3 * BLD floats)
{
  constexpr int AS   = 40;               // LDS row stride in halves (pad vs. 32)
  constexpr int KTS  = DDIM / 32;        // 24 K-steps
  __shared__ half_t sA[2][128 * AS];     // A tile: [row][k]  128 x 32, double-buffered
  __shared__ half_t sB[2][128 * AS];     // B tile: [n][k]    128 x 32, double-buffered

  const int tid  = threadIdx.x;
  const int lane = tid & 31;
  const int wave = tid >> 5;
  const int wm   = wave & 3;             // 32-row strip within 128
  const int wn   = wave >> 2;            // 64-col strip within 128
  const int l16  = lane & 15;
  const int lgrp = lane >> 4;

  const int mtile = blockIdx.y;          // 0..255
  const int ntile = blockIdx.x;          // 0..5
  const int row0  = mtile * 128;
  const int col0  = ntile * 128;

  const int bidx = row0 / LSEQ;
  int t = times[bidx];
  if (PASS == 0) t = (t - 1 < 0) ? 0 : (t - 1);

  const half_t* Arow = A + (size_t)row0 * DDIM;                          // uniform
  const half_t* Brow = Wt + (size_t)t * DDIM * DDIM + (size_t)col0 * DDIM; // uniform

  // staging map: 512 chunks of 8 halves; thread owns chunks tid and tid+256
  unsigned gofs[2];                      // global byte offset (sans kk)
  unsigned ldsA[2][2], ldsB[2][2];       // [buf][c2] LDS byte addresses
#pragma unroll
  for (int c2 = 0; c2 < 2; ++c2) {
    int c  = tid + c2 * 256;
    int r  = c >> 2;                     // 0..127 (row of A tile / n of B tile)
    int k8 = (c & 3) * 8;                // 0,8,16,24
    gofs[c2] = (unsigned)((r * DDIM + k8) * 2);
#pragma unroll
    for (int buf = 0; buf < 2; ++buf) {
      ldsA[buf][c2] = (unsigned)(size_t)&sA[buf][r * AS + k8];
      ldsB[buf][c2] = (unsigned)(size_t)&sB[buf][r * AS + k8];
    }
  }

  auto stage_async = [&](int buf, int kk) {
    const unsigned koff = (unsigned)(kk * 2);
#pragma unroll
    for (int c2 = 0; c2 < 2; ++c2) {
      async_copy_b128(ldsA[buf][c2], gofs[c2] + koff, Arow);
      async_copy_b128(ldsB[buf][c2], gofs[c2] + koff, Brow);
    }
  };

  v8f acc[2][4];
#pragma unroll
  for (int i = 0; i < 2; ++i)
#pragma unroll
    for (int j = 0; j < 4; ++j) acc[i][j] = v8f{};

  // prologue: DMA tile 0 into buffer 0
  stage_async(0, 0);
  wait_asynccnt0();
  __syncthreads();

  for (int ki = 0; ki < KTS; ++ki) {
    const int cur = ki & 1;
    if (ki + 1 < KTS) stage_async(cur ^ 1, (ki + 1) * 32);  // DMA next tile, overlap with math

    // fragment loads per documented CDNA5 WMMA VGPR layouts
    v16h afrag[2], bfrag[4];
#pragma unroll
    for (int mt = 0; mt < 2; ++mt) {
      int r  = wm * 32 + mt * 16 + l16;
      int kb = lgrp * 8;                 // lanes 0-15: K0-7,16-23 ; lanes 16-31: K8-15,24-31
      union { v16h h; uint4 q[2]; } u;
      u.q[0] = *(const uint4*)(&sA[cur][r * AS + kb]);
      u.q[1] = *(const uint4*)(&sA[cur][r * AS + kb + 16]);
      afrag[mt] = u.h;
    }
#pragma unroll
    for (int nt = 0; nt < 4; ++nt) {
      int n  = wn * 64 + nt * 16 + l16;
      int kb = lgrp * 8;                 // lanes 0-15: K0-7,16-23 ; lanes 16-31: K8-15,24-31
      union { v16h h; uint4 q[2]; } u;
      u.q[0] = *(const uint4*)(&sB[cur][n * AS + kb]);
      u.q[1] = *(const uint4*)(&sB[cur][n * AS + kb + 16]);
      bfrag[nt] = u.h;
    }

#pragma unroll
    for (int mt = 0; mt < 2; ++mt)
#pragma unroll
      for (int nt = 0; nt < 4; ++nt)
        acc[mt][nt] = __builtin_amdgcn_wmma_f32_16x16x32_f16(
            false, afrag[mt], false, bfrag[nt], (short)0, acc[mt][nt], false, false);

    if (ki + 1 < KTS) {
      wait_asynccnt0();                  // own DMA done; barrier makes it visible to all
      __syncthreads();
    }
  }

  // ---------------- epilogue ----------------
  const float* bb = bias + (size_t)t * DDIM;
  float bv[4];
#pragma unroll
  for (int nt = 0; nt < 4; ++nt) bv[nt] = bb[col0 + wn * 64 + nt * 16 + l16];

#pragma unroll
  for (int mt = 0; mt < 2; ++mt) {
#pragma unroll
    for (int v = 0; v < 8; ++v) {
      // D layout: VGPR v, lane l -> M = v + 8*(l>>4), N = l&15
      int row = row0 + wm * 32 + mt * 16 + v + 8 * lgrp;
      if (STAGE == 1) {
#pragma unroll
        for (int nt = 0; nt < 4; ++nt) {
          int col = col0 + wn * 64 + nt * 16 + l16;
          float x = acc[mt][nt][v] + bv[nt];
          Hout[(size_t)row * DDIM + col] = (half_t)fast_tanh(x);
        }
      } else {
        float f = filt[row];
        int tok = 0;
        if (PASS == 1) tok = reviews[row];
#pragma unroll
        for (int nt = 0; nt < 4; ++nt) {
          int col = col0 + wn * 64 + nt * 16 + l16;
          float y = (acc[mt][nt][v] + bv[nt]) * f;
          Out[(size_t)PASS * BLD + (size_t)row * DDIM + col] = y;
          if (PASS == 1) {
            float e = emb[(size_t)tok * DDIM + col];
            Out[2 * BLD + (size_t)row * DDIM + col] = e + y;
          }
        }
      }
    }
  }
}

// ---------------- launcher ----------------
extern "C" void kernel_launch(void* const* d_in, const int* in_sizes, int n_in,
                              void* d_out, int out_size, void* d_ws, size_t ws_size,
                              hipStream_t stream) {
  (void)in_sizes; (void)n_in; (void)out_size; (void)ws_size;

  const int*   reviews = (const int*)d_in[0];    // [64, 512]
  const int*   times   = (const int*)d_in[1];    // [64]
  const int*   vf      = (const int*)d_in[2];    // [5000]
  const float* emb     = (const float*)d_in[3];  // [30522, 768]
  const float* W1      = (const float*)d_in[4];  // [12, 768, 768]
  const float* b1      = (const float*)d_in[5];  // [12, 768]
  const float* W2      = (const float*)d_in[6];  // [12, 768, 768]
  const float* b2      = (const float*)d_in[7];  // [12, 768]
  float* out = (float*)d_out;

  // workspace layout (bytes)
  char* ws = (char*)d_ws;
  half_t*       Xh   = (half_t*)(ws + 0);               // 50,331,648 B
  half_t*       Hh   = (half_t*)(ws + 50331648);        // 50,331,648 B
  half_t*       W1t  = (half_t*)(ws + 100663296);       // 14,155,776 B
  half_t*       W2t  = (half_t*)(ws + 114819072);       // 14,155,776 B
  float*        filt = (float*)(ws + 128974848);        //    131,072 B
  unsigned int* bm   = (unsigned int*)(ws + 129105920); //      4,096 B

  k_zero_bitmap<<<4, 256, 0, stream>>>(bm);
  k_scatter_filter<<<(NFILT + 255) / 256, 256, 0, stream>>>(vf, bm);
  k_row_filter<<<NROWS / 256, 256, 0, stream>>>(reviews, bm, filt);
  {
    dim3 g(DDIM / 32, DDIM / 32, 2 * NTIME);
    dim3 b(32, 8);
    k_transpose_weights<<<g, b, 0, stream>>>(W1, W2, W1t, W2t);
  }
  k_gather_embs<<<(unsigned)(BLD / 4 / 256), 256, 0, stream>>>(reviews, emb, Xh);

  dim3 grid(DDIM / 128, NROWS / 128);  // (6, 256)
  // pass 0: t_last = max(times-1, 0) -> offset_last
  gemm_kernel<1, 0><<<grid, 256, 0, stream>>>(Xh, W1t, b1, times, filt, reviews, emb, Hh, out);
  gemm_kernel<2, 0><<<grid, 256, 0, stream>>>(Hh, W2t, b2, times, filt, reviews, emb, Hh, out);
  // pass 1: t = times -> offset_now + input_embs
  gemm_kernel<1, 1><<<grid, 256, 0, stream>>>(Xh, W1t, b1, times, filt, reviews, emb, Hh, out);
  gemm_kernel<2, 1><<<grid, 256, 0, stream>>>(Hh, W2t, b2, times, filt, reviews, emb, Hh, out);
}
